// Trimodal_CMD_27900107555223
// MI455X (gfx1250) — matrix operand order repaired and checked by reading, not compiled, not run on hardware
//
#include <hip/hip_runtime.h>
#include <hip/hip_bf16.h>
#include <math.h>

typedef __attribute__((ext_vector_type(16))) __bf16 v16bf;
typedef __attribute__((ext_vector_type(8)))  __bf16 v8bf;
typedef __attribute__((ext_vector_type(8)))  float  v8f;
typedef __attribute__((ext_vector_type(4)))  float  v4f;

#define NH     6
#define DK     13
#define DKP    16
#define DMODEL 78
#define DPAD   96            // padded model dim (multiple of 32)
#define WROWS  80            // padded weight out-cols (multiple of 16)
#define SEQ    1024
#define BATCH  8
#define NROWS  4099          // 4*1024 + 3
#define ATT_SCALE 0.2773500981126146f   // 1/sqrt(13)

#define CAT16(lo, hi) __builtin_shufflevector((lo), (hi), 0,1,2,3,4,5,6,7,8,9,10,11,12,13,14,15)

// ---------------------------------------------------------------------------
// Pad/convert input: X (rows x 78 f32) -> Xp (rows x 96 bf16), zero-padded K.
// ---------------------------------------------------------------------------
__global__ void pad_input_kernel(const float* __restrict__ X, __bf16* __restrict__ Xp,
                                 int rows) {
    int idx = blockIdx.x * blockDim.x + threadIdx.x;
    if (idx >= rows * DPAD) return;
    int r = idx / DPAD, d = idx - r * DPAD;
    float v = (d < DMODEL) ? X[(size_t)r * DMODEL + d] : 0.f;
    Xp[idx] = (__bf16)v;
}

// ---------------------------------------------------------------------------
// Pad/convert/transpose weight: W (78x78, [k][col]) -> Wt (80x96 bf16, [col][k]).
// ---------------------------------------------------------------------------
__global__ void pad_weight_kernel(const float* __restrict__ W, __bf16* __restrict__ Wt) {
    int idx = blockIdx.x * blockDim.x + threadIdx.x;
    if (idx >= WROWS * DPAD) return;
    int c = idx / DPAD, k = idx - c * DPAD;
    float v = (c < DMODEL && k < DMODEL) ? W[(size_t)k * DMODEL + c] : 0.f;
    Wt[idx] = (__bf16)v;
}

// ---------------------------------------------------------------------------
// Projection: out = Xp(8192x96 bf16) @ Wt^T, one wave per 16x16 tile, 3 WMMAs.
// Store per-head padded bf16 scaled by `scale` (ATT_SCALE for Q folds the
// softmax scaling into the projection): normal [bh][n][16], or transposed
// [bh][16][n] (for V, contiguous PV B-fragments). Optional unscaled f32 copy.
// ---------------------------------------------------------------------------
__global__ __launch_bounds__(32)
void proj_kernel(const __bf16* __restrict__ Xp, const __bf16* __restrict__ Wt,
                 __bf16* __restrict__ outP, float* __restrict__ outF,
                 int vtrans, float scale) {
    const int lane  = threadIdx.x;
    const int l15   = lane & 15;
    const int kgrp  = lane >> 4;
    const int mtile = blockIdx.x;                 // 0..511
    const int ntile = blockIdx.y;                 // 0..4
    const int arow  = mtile * 16 + l15;           // A row (M = lane&15)
    const int bcol  = ntile * 16 + l15;           // B col (N = lane&15), < 80

    const v8bf* Arow = (const v8bf*)(Xp + (size_t)arow * DPAD);
    const v8bf* Brow = (const v8bf*)(Wt + (size_t)bcol * DPAD);

    v8f acc = {};
#pragma unroll
    for (int kt = 0; kt < 3; ++kt) {
        // A 16x32: lane-group 0 holds K={0..7,16..23}, group 1 K={8..15,24..31}
        v8bf a0 = Arow[kt * 4 + kgrp];            // k = kt*32 + kgrp*8 + 0..7
        v8bf a1 = Arow[kt * 4 + 2 + kgrp];        // k = kt*32 + 16 + kgrp*8 + 0..7
        v16bf a = CAT16(a0, a1);
        // B 32x16: group 0 holds K={0..7,16..23}, group 1 K={8..15,24..31}
        v8bf b0 = Brow[kt * 4 + kgrp];
        v8bf b1 = Brow[kt * 4 + 2 + kgrp];
        v16bf b = CAT16(b0, b1);
        acc = __builtin_amdgcn_wmma_f32_16x16x32_bf16(false, a, false, b,
                                                      (short)0, acc, false, false);
    }

    if (bcol < DMODEL) {
        const int h = bcol / DK;
        const int d = bcol - h * DK;
#pragma unroll
        for (int v = 0; v < 8; ++v) {
            int m    = v + kgrp * 8;              // D row = vgpr + 8*lanegroup
            int grow = mtile * 16 + m;
            int bidx = grow >> 10;
            int nrow = grow & 1023;
            size_t bh = (size_t)bidx * NH + h;
            size_t idx = vtrans ? (bh * DKP + d) * SEQ + nrow
                                : (bh * SEQ + nrow) * DKP + d;
            outP[idx] = (__bf16)(acc[v] * scale);
            if (outF) outF[(bh * SEQ + nrow) * DKP + d] = acc[v];
        }
    }
}

// ---------------------------------------------------------------------------
// Row softmax over 16 x SEQ f32 block in LDS: 2 lanes/row, float4 vectorized,
// pair-reduce with shfl_xor(1). Stable (max subtraction), matches jax.
// ---------------------------------------------------------------------------
__device__ __forceinline__ void softmax_rows(float* S, int lane) {
    const int r = lane >> 1, half = lane & 1;
    v4f* row = (v4f*)(S + (size_t)r * SEQ + half * (SEQ / 2));
    float m = -3.4e38f;
    for (int i = 0; i < SEQ / 8; ++i) {
        v4f t = row[i];
        m = fmaxf(m, fmaxf(fmaxf(t[0], t[1]), fmaxf(t[2], t[3])));
    }
    m = fmaxf(m, __shfl_xor(m, 1, 32));
    float s = 0.f;
    for (int i = 0; i < SEQ / 8; ++i) {
        v4f t = row[i];
        t[0] = __expf(t[0] - m); t[1] = __expf(t[1] - m);
        t[2] = __expf(t[2] - m); t[3] = __expf(t[3] - m);
        row[i] = t;
        s += t[0] + t[1] + t[2] + t[3];
    }
    s += __shfl_xor(s, 1, 32);
    float inv = 1.0f / s;
    for (int i = 0; i < SEQ / 8; ++i) row[i] *= inv;
}

// ---------------------------------------------------------------------------
// Attention: one wave per (b,h,qtile). Scores (16x1024 f32 = 64KB) live in
// LDS; 64 score WMMAs + 32 PV WMMAs. contrastive: softmax -> 1-p -> softmax,
// plus f32 q residual. V pre-transposed; Q pre-scaled by 1/sqrt(dk).
// Lane-group 1's score B-fragment (K rows 16..31, structurally zero) is
// loaded from a zeroed scratch row -> no per-iteration vector selects.
// ---------------------------------------------------------------------------
__global__ __launch_bounds__(32)
void attn_kernel(const __bf16* __restrict__ Q, const __bf16* __restrict__ K,
                 const __bf16* __restrict__ Vt, const float* __restrict__ QF,
                 const __bf16* __restrict__ zrow,
                 float* __restrict__ out, int row_off, int contrastive) {
    __shared__ float S[16 * SEQ];

    const int lane  = threadIdx.x;
    const int l15   = lane & 15;
    const int kgrp  = lane >> 4;
    const int qtile = blockIdx.x & 63;
    const int bh    = blockIdx.x >> 6;
    const int b     = bh / NH;
    const int h     = bh - b * NH;

    const __bf16* Qb  = Q  + (size_t)bh * SEQ * DKP;
    const __bf16* Kb  = K  + (size_t)bh * SEQ * DKP;
    const __bf16* Vtb = Vt + (size_t)bh * DKP * SEQ;

    const v8bf z8 = {};

    // Q A-fragment: group 0 holds k = 0..15 (dk<=16), group 1 (k 16..31) zero.
    const int qrow = qtile * 16 + l15;
    const v8bf* Qrow = (const v8bf*)(Qb + (size_t)qrow * DKP);
    v16bf aq = kgrp ? CAT16(z8, z8) : CAT16(Qrow[0], Qrow[1]);

    // group-1 lanes read their (all-zero) B rows from the zero scratch row
    const __bf16* kbase = kgrp ? zrow : Kb;

    // scores: S[16][SEQ] = (q*scale) @ k^T   (scale pre-folded into Q)
    for (int t = 0; t < SEQ / 16; ++t) {
        const int key = t * 16 + l15;
        const size_t koff = kgrp ? 0 : (size_t)key * DKP;
        const v8bf* Krow = (const v8bf*)(kbase + koff);
        v8bf k0 = Krow[0], k1 = Krow[1];
        v16bf bk = CAT16(k0, k1);
        v8f c = {};
        c = __builtin_amdgcn_wmma_f32_16x16x32_bf16(false, aq, false, bk,
                                                    (short)0, c, false, false);
#pragma unroll
        for (int v = 0; v < 8; ++v) {
            int m = v + kgrp * 8;
            S[(size_t)m * SEQ + t * 16 + l15] = c[v];
        }
    }
    __syncthreads();

    softmax_rows(S, lane);
    if (contrastive) {
        const int r = lane >> 1, half = lane & 1;
        v4f* row = (v4f*)(S + (size_t)r * SEQ + half * (SEQ / 2));
        for (int i = 0; i < SEQ / 8; ++i) {
            v4f t = row[i];
            t[0] = 1.f - t[0]; t[1] = 1.f - t[1];
            t[2] = 1.f - t[2]; t[3] = 1.f - t[3];
            row[i] = t;
        }
        __syncthreads();
        softmax_rows(S, lane);
    }
    __syncthreads();

    // PV: acc(16x16) = sum over key chunks of P(16x32) @ V(32x16)
    v8f acc = {};
    for (int c32 = 0; c32 < SEQ / 32; ++c32) {
        // A: probs, lane-group pattern k = {kgrp*8+0..7, 16+kgrp*8+0..7}
        const v4f* Prow = (const v4f*)(S + (size_t)l15 * SEQ + c32 * 32 + kgrp * 8);
        v4f p0 = Prow[0], p1 = Prow[1];           // first 8
        v4f p2 = Prow[4], p3 = Prow[5];           // +16 floats
        v16bf ap;
#pragma unroll
        for (int j = 0; j < 4; ++j) {
            ap[j]      = (__bf16)p0[j];
            ap[4 + j]  = (__bf16)p1[j];
            ap[8 + j]  = (__bf16)p2[j];
            ap[12 + j] = (__bf16)p3[j];
        }
        // B: Vt row l15, keys c32*32 + {kgrp*8+0..7, 16+kgrp*8+0..7}
        const v8bf* Vrow = (const v8bf*)(Vtb + (size_t)l15 * SEQ);
        v8bf b0 = Vrow[c32 * 4 + kgrp];
        v8bf b1 = Vrow[c32 * 4 + 2 + kgrp];
        v16bf bv = CAT16(b0, b1);
        acc = __builtin_amdgcn_wmma_f32_16x16x32_bf16(false, ap, false, bv,
                                                      (short)0, acc, false, false);
    }

    const int d = l15;
    if (d < DK) {
#pragma unroll
        for (int v = 0; v < 8; ++v) {
            int m    = v + kgrp * 8;
            int orow = qtile * 16 + m;
            float val = acc[v];
            if (QF) val += QF[((size_t)bh * SEQ + orow) * DKP + d];
            out[((size_t)b * NROWS + row_off + orow) * DMODEL + h * DK + d] = val;
        }
    }
}

// ---------------------------------------------------------------------------
// Pooled tokens: tok0 = mean(x), tok1 = mean(att_tv+c_tv), tok2 = mean(att_ta+c_ta)
// ---------------------------------------------------------------------------
__global__ void mean_kernel(const float* __restrict__ x, const float* __restrict__ out,
                            float* __restrict__ tokens) {
    int idx = blockIdx.x * blockDim.x + threadIdx.x;
    if (idx >= BATCH * 3 * DMODEL) return;
    int d = idx % DMODEL;
    int t = (idx / DMODEL) % 3;
    int b = idx / (3 * DMODEL);
    float s = 0.f;
    if (t == 0) {
        const float* p = x + (size_t)b * SEQ * DMODEL + d;
        for (int n = 0; n < SEQ; ++n) s += p[(size_t)n * DMODEL];
    } else {
        size_t base = (size_t)b * NROWS * DMODEL;
        size_t r0   = (t == 1) ? 0 : 2048;
        const float* p0 = out + base + r0 * DMODEL + d;
        const float* p1 = out + base + (r0 + 1024) * DMODEL + d;
        for (int n = 0; n < SEQ; ++n)
            s += p0[(size_t)n * DMODEL] + p1[(size_t)n * DMODEL];
    }
    tokens[idx] = s * (1.0f / SEQ);
}

// ---------------------------------------------------------------------------
// 3-token MHSA on pooled tokens (tiny): one thread per (b,h), scalar f32.
// ---------------------------------------------------------------------------
__global__ __launch_bounds__(64)
void fused_kernel(const float* __restrict__ tokens,
                  const float* __restrict__ Wq, const float* __restrict__ Wk,
                  const float* __restrict__ Wv, float* __restrict__ out) {
    int t = threadIdx.x;
    if (t >= BATCH * NH) return;
    int b = t / NH, h = t % NH;
    float q[3][DK], k[3][DK], v[3][DK];
    for (int i = 0; i < 3; ++i) {
        const float* tok = tokens + ((size_t)b * 3 + i) * DMODEL;
        for (int d = 0; d < DK; ++d) {
            int c0 = h * DK + d;
            float sq = 0.f, sk = 0.f, sv = 0.f;
            for (int c = 0; c < DMODEL; ++c) {
                float tv = tok[c];
                sq += tv * Wq[(size_t)c * DMODEL + c0];
                sk += tv * Wk[(size_t)c * DMODEL + c0];
                sv += tv * Wv[(size_t)c * DMODEL + c0];
            }
            q[i][d] = sq; k[i][d] = sk; v[i][d] = sv;
        }
    }
    for (int i = 0; i < 3; ++i) {
        float s[3];
        float mx = -3.4e38f;
        for (int j = 0; j < 3; ++j) {
            float acc = 0.f;
            for (int d = 0; d < DK; ++d) acc += q[i][d] * k[j][d];
            s[j] = acc * ATT_SCALE;
            mx = fmaxf(mx, s[j]);
        }
        float sum = 0.f;
        for (int j = 0; j < 3; ++j) { s[j] = __expf(s[j] - mx); sum += s[j]; }
        float inv = 1.f / sum;
        for (int d = 0; d < DK; ++d) {
            float o = 0.f;
            for (int j = 0; j < 3; ++j) o += s[j] * inv * v[j][d];
            out[((size_t)b * NROWS + 4096 + i) * DMODEL + h * DK + d] = o;
        }
    }
}

// ---------------------------------------------------------------------------
extern "C" void kernel_launch(void* const* d_in, const int* in_sizes, int n_in,
                              void* d_out, int out_size, void* d_ws, size_t ws_size,
                              hipStream_t stream) {
    (void)in_sizes; (void)n_in; (void)out_size; (void)ws_size;
    const float* x = (const float*)d_in[0];
    const float* y = (const float*)d_in[1];
    const float* z = (const float*)d_in[2];
    const float* W[15];
    for (int i = 0; i < 15; ++i) W[i] = (const float*)d_in[5 + i];
    float* out = (float*)d_out;

    const size_t E  = (size_t)BATCH * NH * SEQ * DKP;   // 786432 elems
    const size_t EP = (size_t)BATCH * SEQ * DPAD;       // padded input elems
    const size_t EW = (size_t)WROWS * DPAD;             // padded weight elems

    char* ws = (char*)d_ws;
    size_t off = 0;
    __bf16* bufs[12];                                   // Q,K,Vt for 4 blocks
    for (int i = 0; i < 12; ++i) { bufs[i] = (__bf16*)(ws + off); off += E * sizeof(__bf16); }
    __bf16* zrow = (__bf16*)(ws + off); off += 64;      // 32B zero row (+pad)
    size_t qkv_bytes = off;                             // memset region
    float* qf_ctv = (float*)(ws + off); off += E * sizeof(float);
    float* qf_cta = (float*)(ws + off); off += E * sizeof(float);
    float* tokens = (float*)(ws + off); off += (size_t)BATCH * 3 * DMODEL * sizeof(float);
    __bf16* xp = (__bf16*)(ws + off); off += EP * sizeof(__bf16);
    __bf16* yp = (__bf16*)(ws + off); off += EP * sizeof(__bf16);
    __bf16* zp = (__bf16*)(ws + off); off += EP * sizeof(__bf16);
    __bf16* wt[12];
    for (int i = 0; i < 12; ++i) { wt[i] = (__bf16*)(ws + off); off += EW * sizeof(__bf16); }

    // zero QKV region (K pad cols + zero row must read as 0.0)
    hipMemsetAsync(d_ws, 0, qkv_bytes, stream);

    // pad/convert inputs and weights to bf16 WMMA layouts
    {
        int rows = BATCH * SEQ;
        int nthr = rows * DPAD;
        pad_input_kernel<<<dim3((nthr + 255) / 256), dim3(256), 0, stream>>>(x, xp, rows);
        pad_input_kernel<<<dim3((nthr + 255) / 256), dim3(256), 0, stream>>>(y, yp, rows);
        pad_input_kernel<<<dim3((nthr + 255) / 256), dim3(256), 0, stream>>>(z, zp, rows);
        int wthr = (int)EW;
        // weight usage order: Q,K,V for blocks tv, ctv, ta, cta
        const int worder[12] = {0,1,2, 6,7,8, 3,4,5, 9,10,11};
        for (int i = 0; i < 12; ++i)
            pad_weight_kernel<<<dim3((wthr + 255) / 256), dim3(256), 0, stream>>>(W[worder[i]], wt[i]);
    }

    // output row order: att_tv | c_tv | att_ta | c_ta | fused
    const __bf16* Yin[4] = { yp, yp, zp, zp };
    const int contr[4]   = { 0, 1, 0, 1 };
    const int roff[4]    = { 0, 1024, 2048, 3072 };
    float*    qfp[4]     = { nullptr, qf_ctv, nullptr, qf_cta };

    dim3 pgrid(512, 5), pblk(32);
    for (int blk = 0; blk < 4; ++blk) {
        // Q: bf16 copy pre-scaled by 1/sqrt(dk); f32 residual copy unscaled
        proj_kernel<<<pgrid, pblk, 0, stream>>>(xp,       wt[blk * 3 + 0], bufs[blk * 3 + 0], qfp[blk], 0, ATT_SCALE);
        proj_kernel<<<pgrid, pblk, 0, stream>>>(Yin[blk], wt[blk * 3 + 1], bufs[blk * 3 + 1], nullptr,  0, 1.0f);
        proj_kernel<<<pgrid, pblk, 0, stream>>>(Yin[blk], wt[blk * 3 + 2], bufs[blk * 3 + 2], nullptr,  1, 1.0f);
    }
    for (int blk = 0; blk < 4; ++blk) {
        attn_kernel<<<dim3(BATCH * NH * (SEQ / 16)), dim3(32), 0, stream>>>(
            bufs[blk * 3 + 0], bufs[blk * 3 + 1], bufs[blk * 3 + 2],
            qfp[blk], zrow, out, roff[blk], contr[blk]);
    }
    mean_kernel<<<dim3((BATCH * 3 * DMODEL + 255) / 256), dim3(256), 0, stream>>>(x, out, tokens);
    fused_kernel<<<dim3(1), dim3(64), 0, stream>>>(tokens, W[12], W[13], W[14], out);
}
